// NGCF_52862457479750
// MI455X (gfx1250) — matrix operand order repaired and checked by reading, not compiled, not run on hardware
//
#include <hip/hip_runtime.h>

#define D    64
#define OUTD 256

typedef __attribute__((ext_vector_type(2)))  float    v2f;
typedef __attribute__((ext_vector_type(8)))  float    v8f;
typedef __attribute__((ext_vector_type(16))) _Float16 v16h;

// ---------------------------------------------------------------- utilities
__global__ void zero_kernel(float* __restrict__ p, long long n) {
    long long i = (long long)blockIdx.x * blockDim.x + threadIdx.x;
    long long s = (long long)gridDim.x * blockDim.x;
    for (; i < n; i += s) p[i] = 0.0f;
}

// deg[c] += 1 per edge (float counts are exact up to 2^24; E = 1e6)
__global__ void deg_kernel(const int* __restrict__ ei, float* __restrict__ deg, int E) {
    int e = blockIdx.x * blockDim.x + threadIdx.x;
    if (e < E) atomicAdd(&deg[ei[E + e]], 1.0f);
}

// in-place deg -> dinv
__global__ void dinv_kernel(float* __restrict__ dinv, int N) {
    int i = blockIdx.x * blockDim.x + threadIdx.x;
    if (i < N) {
        float d = dinv[i];
        dinv[i] = (d > 0.0f) ? rsqrtf(fmaxf(d, 1.0f)) : 0.0f;
    }
}

// s0[r] += dinv[r]*dinv[c]  (layer-independent norm sums for the bias term)
__global__ void normsum_kernel(const int* __restrict__ ei, const float* __restrict__ dinv,
                               float* __restrict__ s0, int E) {
    int e = blockIdx.x * blockDim.x + threadIdx.x;
    if (e < E) {
        int r = ei[e], c = ei[E + e];
        atomicAdd(&s0[r], dinv[r] * dinv[c]);
    }
}

// emb[N][64] -> out[N][256] columns 0..63
__global__ void copy_emb_kernel(const float* __restrict__ emb, float* __restrict__ out, int N) {
    int t = blockIdx.x * blockDim.x + threadIdx.x;
    int n = t >> 4, q = t & 15;
    if (n < N) {
        const float4* src = (const float4*)(emb + (size_t)n * D);
        float4*       dst = (float4*)(out + (size_t)n * OUTD);
        dst[q] = src[q];
    }
}

// ------------------------------------------------------- edge scatter phase
// One wave per edge; lane handles dims {2*lane, 2*lane+1}.
//   S1[r] += norm * x[c] ;  S2[r] += norm * (x[r] .* x[c])
__global__ void scatter_kernel(const int* __restrict__ ei,
                               const float* __restrict__ x, int ldx,
                               const float* __restrict__ dinv,
                               float* __restrict__ S1, float* __restrict__ S2, int E) {
    int t    = blockIdx.x * blockDim.x + threadIdx.x;
    int e    = t >> 5;
    int lane = t & 31;
    if (e >= E) return;
    int r = ei[e], c = ei[E + e];
    float nrm = dinv[r] * dinv[c];
    const float2 xc = *(const float2*)(x + (size_t)c * ldx + 2 * lane);
    const float2 xr = *(const float2*)(x + (size_t)r * ldx + 2 * lane);
    float* s1p = S1 + (size_t)r * D + 2 * lane;
    float* s2p = S2 + (size_t)r * D + 2 * lane;
    atomicAdd(s1p + 0, nrm * xc.x);
    atomicAdd(s1p + 1, nrm * xc.y);
    atomicAdd(s2p + 0, nrm * xr.x * xc.x);
    atomicAdd(s2p + 1, nrm * xr.y * xc.y);
}

// ------------------------------------------------- node transform via WMMA
// Block = 256 threads = 8 waves = 2 node-tiles of 16 rows; each wave owns a
// 16x16 output tile (4 column-tiles per node-tile).
//   Y = S1 @ W1^T + S2 @ W2^T + s0 * (b1+b2), then LeakyReLU(0.2).
__global__ __launch_bounds__(256) void transform_kernel(
        const float* __restrict__ S1, const float* __restrict__ S2,
        const float* __restrict__ s0,
        const float* __restrict__ W1, const float* __restrict__ W2,
        const float* __restrict__ b1, const float* __restrict__ b2,
        float* __restrict__ xout /* row stride OUTD */) {
    int wv   = threadIdx.x >> 5;
    int lane = threadIdx.x & 31;
    int tile = blockIdx.x * 2 + (wv >> 2);   // 16-row node tile
    int ct   = wv & 3;                       // 16-col column tile
    int base = tile * 16;
    int m    = lane & 15;
    int half = lane >> 4;
    int n    = ct * 16 + m;                  // output column in [0,64)

    v8f acc = {};

#if __has_builtin(__builtin_amdgcn_wmma_f32_16x16x4_f32)
    // f32 WMMA: exact fp32 math. A[m][k0+2*half+i], B[k][n] = W[n][k].
    const float* a1p = S1 + (size_t)(base + m) * D + 2 * half;
    const float* a2p = S2 + (size_t)(base + m) * D + 2 * half;
    const float* w1p = W1 + (size_t)n * D + 2 * half;
    const float* w2p = W2 + (size_t)n * D + 2 * half;
    for (int k0 = 0; k0 < D; k0 += 4) {
        v2f a1  = { a1p[k0], a1p[k0 + 1] };
        v2f bw1 = { w1p[k0], w1p[k0 + 1] };
        acc = __builtin_amdgcn_wmma_f32_16x16x4_f32(false, a1, false, bw1,
                                                    (short)0, acc, false, false);
        v2f a2  = { a2p[k0], a2p[k0 + 1] };
        v2f bw2 = { w2p[k0], w2p[k0 + 1] };
        acc = __builtin_amdgcn_wmma_f32_16x16x4_f32(false, a2, false, bw2,
                                                    (short)0, acc, false, false);
    }
#else
    // Fallback: probe-confirmed f16 WMMA (f32 accumulate).
    const float* a1row = S1 + (size_t)(base + m) * D;
    const float* a2row = S2 + (size_t)(base + m) * D;
    const float* w1row = W1 + (size_t)n * D;
    const float* w2row = W2 + (size_t)n * D;
    for (int k0 = 0; k0 < D; k0 += 32) {
        v16h a1, a2, bw1, bw2;
#pragma unroll
        for (int v = 0; v < 8; ++v) {
            int ka = (v < 4) ? (v * 2 + 8 * half) : (16 + (v - 4) * 2 + 8 * half);
            a1[2 * v]     = (_Float16)a1row[k0 + ka];
            a1[2 * v + 1] = (_Float16)a1row[k0 + ka + 1];
            a2[2 * v]     = (_Float16)a2row[k0 + ka];
            a2[2 * v + 1] = (_Float16)a2row[k0 + ka + 1];
            int kb = (v < 4) ? (v * 2 + 8 * half) : (16 + (v - 4) * 2 + 8 * half);
            bw1[2 * v]     = (_Float16)w1row[k0 + kb];
            bw1[2 * v + 1] = (_Float16)w1row[k0 + kb + 1];
            bw2[2 * v]     = (_Float16)w2row[k0 + kb];
            bw2[2 * v + 1] = (_Float16)w2row[k0 + kb + 1];
        }
        acc = __builtin_amdgcn_wmma_f32_16x16x32_f16(false, a1, false, bw1,
                                                     (short)0, acc, false, false);
        acc = __builtin_amdgcn_wmma_f32_16x16x32_f16(false, a2, false, bw2,
                                                     (short)0, acc, false, false);
    }
#endif

    float bsum = b1[n] + b2[n];
#pragma unroll
    for (int v = 0; v < 8; ++v) {
        int mm  = base + v + 8 * half;       // C/D layout: VGPR v -> M = v + 8*half
        float y = acc[v] + s0[mm] * bsum;
        y = (y >= 0.0f) ? y : 0.2f * y;
        xout[(size_t)mm * OUTD + n] = y;
    }
}

// ------------------------------------------------------------------- driver
extern "C" void kernel_launch(void* const* d_in, const int* in_sizes, int n_in,
                              void* d_out, int out_size, void* d_ws, size_t ws_size,
                              hipStream_t stream) {
    const int*   ei  = (const int*)d_in[0];
    const float* emb = (const float*)d_in[1];
    const float* W1  = (const float*)d_in[2];
    const float* b1  = (const float*)d_in[3];
    const float* W2  = (const float*)d_in[4];
    const float* b2  = (const float*)d_in[5];
    float*       out = (float*)d_out;

    const int E = in_sizes[0] / 2;
    const int N = in_sizes[1] / D;
    const int L = in_sizes[2] / (D * D);

    float* dinv = (float*)d_ws;            // [N]
    float* s0   = dinv + N;                // [N]
    float* S1   = s0 + N;                  // [N*D]
    float* S2   = S1 + (size_t)N * D;      // [N*D]

    // Degree normalization (computed once; layer independent)
    zero_kernel<<<1024, 256, 0, stream>>>(dinv, 2LL * N);            // dinv & s0
    deg_kernel<<<(E + 255) / 256, 256, 0, stream>>>(ei, dinv, E);
    dinv_kernel<<<(N + 255) / 256, 256, 0, stream>>>(dinv, N);
    normsum_kernel<<<(E + 255) / 256, 256, 0, stream>>>(ei, dinv, s0, E);

    // Layer-0 embedding -> out columns [0,64)
    copy_emb_kernel<<<(N * 16 + 255) / 256, 256, 0, stream>>>(emb, out, N);

    const int tblocks = N / 32;                                  // 3750, exact
    const long long sthreads = (long long)E * 32;
    const int sblocks = (int)((sthreads + 255) / 256);

    for (int l = 0; l < L; ++l) {
        zero_kernel<<<2048, 256, 0, stream>>>(S1, 2LL * N * D);  // S1 & S2
        scatter_kernel<<<sblocks, 256, 0, stream>>>(
            ei, out + (size_t)l * D, OUTD, dinv, S1, S2, E);
        transform_kernel<<<tblocks, 256, 0, stream>>>(
            S1, S2, s0,
            W1 + (size_t)l * D * D, W2 + (size_t)l * D * D,
            b1 + (size_t)l * D, b2 + (size_t)l * D,
            out + (size_t)(l + 1) * D);
    }
}
